// BothMamba_8718783611647
// MI455X (gfx1250) — matrix operand order repaired and checked by reading, not compiled, not run on hardware
//
#include <hip/hip_runtime.h>
#include <hip/hip_bf16.h>
#include <math.h>

// ---------------------------------------------------------------------------
// BothMamba (SpaMamba + SpeMamba) for MI455X / gfx1250, wave32.
//  - WMMA f16->f32 for the three large projections
//  - TDM (tensor_load_to_lds) for the strided input repack, with TDM LDS
//    padding used to get a conflict-free transpose pitch
//  - chunked associative scan for the L=16384 SSM recurrence
// ---------------------------------------------------------------------------

typedef __attribute__((ext_vector_type(16))) _Float16 v16h;
typedef __attribute__((ext_vector_type(8)))  _Float16 half8;
typedef __attribute__((ext_vector_type(8)))  float    v8f;
typedef __attribute__((ext_vector_type(4)))  unsigned int u32x4;
typedef __attribute__((ext_vector_type(4)))  int          i32x4v;
typedef __attribute__((ext_vector_type(8)))  int          i32x8v;

#define DEVINL __device__ __forceinline__

DEVINL float silu_f(float v)     { return v / (1.f + __expf(-v)); }
DEVINL float softplus_f(float v) { return (v > 20.f) ? v : log1pf(__expf(v)); }

// Problem dims (fixed by the reference).
constexpr int NTOK = 16384;        // B*H*W tokens for SpaMamba
constexpr int CCH  = 64;           // channels / d_model (spa)
constexpr int DI   = 128;          // spa d_inner
constexpr int DS   = 16;           // d_state
constexpr int CL   = 64;           // scan chunk length
constexpr int NCH  = NTOK / CL;    // 256 chunks
constexpr int NPAIR = DI * DS;     // 2048 (d,s) recurrence pairs

// ---------------------------------------------------------------------------
// K0: repack x [B,C,H,W] -> Xtok [N, C] via the Tensor Data Mover.
// Each block handles 64 tokens (same batch image, contiguous hw) x 64 chans:
// a 2D tile (64 rows of 64 contiguous f32, row stride 4096) DMA'd into LDS.
// TDM pad: +1 DWORD after every 64 DWORDs -> LDS pitch 65 floats, so the
// subsequent transpose reads are bank-conflict free.
// ---------------------------------------------------------------------------
__global__ __launch_bounds__(256)
void repack_x_tdm(const float* __restrict__ x, float* __restrict__ Xtok)
{
  __shared__ __align__(16) float sTile[64 * 65];
  const int tid = threadIdx.x;
  const int n0  = blockIdx.x * 64;
  const int b   = n0 >> 12;
  const int hw0 = n0 & 4095;

  if (tid < 32) {  // wave 0 issues the descriptor (TDM ignores EXEC)
    unsigned long long gaddr =
        (unsigned long long)(const void*)(x + (size_t)b * 262144 + hw0);
    unsigned lds_addr = (unsigned)(unsigned long long)(const void*)&sTile[0];

    // D# group 0: count=1 (user), lds_addr, global_addr[56:0], type=2.
    u32x4 g0;
    g0[0] = 1u;
    g0[1] = lds_addr;
    g0[2] = (unsigned)(gaddr & 0xFFFFFFFFull);
    g0[3] = ((unsigned)((gaddr >> 32) & 0x01FFFFFFull)) | (2u << 30);

    // D# group 1: data_size=4B (code 2), pad_enable=1, pad_interval=5
    // (64 DWORDs), pad_amount=0 (1 DWORD); tensor 64x64, tile 64x64,
    // tensor_dim0_stride = 4096 elements (H*W).
    i32x8v g1;
    g1[0] = (int)((2u << 16) | (1u << 20) | (5u << 22));
    g1[1] = (int)(64u << 16);   // tensor_dim0 = 64      (bits 79:48)
    g1[2] = (int)(64u << 16);   // tensor_dim1 = 64      (bits 111:80)
    g1[3] = (int)(64u << 16);   // tile_dim0   = 64      (bits 127:112)
    g1[4] = 64;                 // tile_dim1   = 64      (bits 143:128)
    g1[5] = 4096;               // tensor_dim0_stride    (bits 207:160)
    g1[6] = 0;
    g1[7] = 0;

    i32x4v zz = {0, 0, 0, 0};   // groups 2/3 unused (2D tensor)
#if defined(__clang_major__) && (__clang_major__ >= 23)
    i32x8v z8 = {0, 0, 0, 0, 0, 0, 0, 0};
    __builtin_amdgcn_tensor_load_to_lds(g0, g1, zz, zz, z8, 0);
#else
    __builtin_amdgcn_tensor_load_to_lds(g0, g1, zz, zz, 0);
#endif
    __builtin_amdgcn_s_wait_tensorcnt(0);
  }
  __syncthreads();

  // Transpose out of LDS: consecutive lanes hit consecutive Xtok addresses
  // (coalesced) and stride-65 LDS rows (conflict-free).
  for (int i = tid; i < 64 * 64; i += 256) {
    int c = i & 63, nl = i >> 6;
    Xtok[(size_t)(n0 + nl) * 64 + c] = sTile[c * 65 + nl];
  }
}

// ---------------------------------------------------------------------------
// WMMA GEMM: Y[N, M] = X[N, K] @ W[M, K]^T  (f32 in/out, f16 WMMA, f32 acc)
// Block = 256 threads (8 waves), 128 tokens/block; wave owns 16 tokens and
// sweeps all M/16 column tiles; K accumulated in steps of 32.
// Fragment layouts per CDNA5 ISA 7.12.2 (wave32).
// ---------------------------------------------------------------------------
template<int K, int M, int MREAL>
__global__ __launch_bounds__(256)
void gemm_xw_t(const float* __restrict__ X, const float* __restrict__ W,
               float* __restrict__ Y)
{
  __shared__ __align__(16) _Float16 sX[128 * K];
  __shared__ __align__(16) _Float16 sW[M * K];
  const int tid = threadIdx.x;
  const int tokBase = blockIdx.x * 128;

  for (int i = tid; i < M * K; i += 256) {
    int m = i / K, k = i - m * K;
    sW[i] = (m < MREAL) ? (_Float16)W[m * K + k] : (_Float16)0.f;
  }
  for (int i = tid; i < 128 * K; i += 256) {
    int r = i / K, k = i - r * K;
    sX[i] = (_Float16)X[(size_t)(tokBase + r) * K + k];
  }
  __syncthreads();

  const int lane = tid & 31;
  const int wave = tid >> 5;
  const int rowTile = wave * 16;
  const int lh = lane & 15;
  const int hi = lane >> 4;        // 0: K 0-7/16-23, 1: K 8-15/24-31 (A frag)

  // A fragments are reused across every N tile -> hoist them.
  v16h afrag[K / 32];
#pragma unroll
  for (int ks = 0; ks < K / 32; ++ks) {
    const half8* pa = (const half8*)&sX[(rowTile + lh) * K + ks * 32 + hi * 8];
    afrag[ks] = __builtin_shufflevector(pa[0], pa[2],
        0,1,2,3,4,5,6,7,8,9,10,11,12,13,14,15);
  }
#pragma unroll 1
  for (int nt = 0; nt < M / 16; ++nt) {
    v8f acc = {};
#pragma unroll
    for (int ks = 0; ks < K / 32; ++ks) {
      // B: column (nt*16+lh), K 0-7/16-23 (hi=0) or 8-15/24-31 (hi=1), same as A.
      const half8* pb = (const half8*)&sW[(nt * 16 + lh) * K + ks * 32 + hi * 8];
      v16h bfrag = __builtin_shufflevector(pb[0], pb[2],
          0,1,2,3,4,5,6,7,8,9,10,11,12,13,14,15);
      acc = __builtin_amdgcn_wmma_f32_16x16x32_f16(
          /*neg_a=*/false, afrag[ks], /*neg_b=*/false, bfrag,
          /*c_mod=*/(short)0, acc, /*reuse_a=*/false, /*reuse_b=*/false);
    }
    // C/D layout: VGPR r -> (M = r + 8*hi, N = lh)
    float* yp = &Y[(size_t)(tokBase + rowTile + hi * 8) * M + nt * 16 + lh];
#pragma unroll
    for (int r = 0; r < 8; ++r) yp[(size_t)r * M] = acc[r];
  }
}

// ---------------------------------------------------------------------------
// K2: causal depthwise conv1d (k=4) over the single fused sequence + SiLU.
// xz[:, 0:128] is xi.
// ---------------------------------------------------------------------------
__global__ void spa_conv_silu(const float* __restrict__ xz,
                              const float* __restrict__ cw,
                              const float* __restrict__ cb,
                              float* __restrict__ xc)
{
  int i = blockIdx.x * blockDim.x + threadIdx.x;   // NTOK*DI
  if (i >= NTOK * DI) return;
  int d = i & (DI - 1);
  int n = i >> 7;
  float acc = cb[d];
#pragma unroll
  for (int j = 0; j < 4; ++j) {
    int t = n - 3 + j;
    if (t >= 0) acc += cw[d * 4 + j] * xz[(size_t)t * 256 + d];
  }
  xc[i] = silu_f(acc);
}

// ---------------------------------------------------------------------------
// K4: dt = softplus(xdb[:, 0:4] @ dt_w^T + dt_b)   (K=4 -> plain VALU)
// ---------------------------------------------------------------------------
__global__ void spa_dt(const float* __restrict__ xdb,
                       const float* __restrict__ dtw,
                       const float* __restrict__ dtb,
                       float* __restrict__ dt)
{
  int i = blockIdx.x * blockDim.x + threadIdx.x;   // NTOK*DI
  if (i >= NTOK * DI) return;
  int d = i & (DI - 1);
  int n = i >> 7;
  float s = dtb[d];
#pragma unroll
  for (int r = 0; r < 4; ++r) s += xdb[(size_t)n * 48 + r] * dtw[d * 4 + r];
  dt[i] = softplus_f(s);
}

// ---------------------------------------------------------------------------
// K5: chunk-local scan aggregates. For each chunk, per (d,s) pair compute the
// affine composition over CL steps: h_end = P*h_start + Q, with
// a_t = exp(dt*A[d][s]), b_t = dt * B_t[s] * xc_t[d]. Thread owns (d, 8 s's).
// ---------------------------------------------------------------------------
__global__ __launch_bounds__(256)
void spa_scan_chunk(const float* __restrict__ dt, const float* __restrict__ xc,
                    const float* __restrict__ xdb, const float* __restrict__ Alog,
                    float* __restrict__ Pg, float* __restrict__ Qg)
{
  __shared__ float sB[16];
  const int tid = threadIdx.x;
  const int d  = tid & (DI - 1);
  const int sb = (tid >> 7) * 8;
  const int chunk = blockIdx.x;
  float Ac[8], P[8], Q[8];
#pragma unroll
  for (int j = 0; j < 8; ++j) {
    Ac[j] = -__expf(Alog[d * DS + sb + j]);
    P[j] = 1.f; Q[j] = 0.f;
  }
  for (int t = 0; t < CL; ++t) {
    int n = chunk * CL + t;
    if (tid < 16) sB[tid] = xdb[(size_t)n * 48 + 4 + tid];
    __syncthreads();
    float dtv = dt[(size_t)n * DI + d];
    float dbx = dtv * xc[(size_t)n * DI + d];
    if (t + 1 < CL) {
      __builtin_prefetch(&dt[(size_t)(n + 1) * DI + d], 0, 1);
      __builtin_prefetch(&xc[(size_t)(n + 1) * DI + d], 0, 1);
    }
#pragma unroll
    for (int j = 0; j < 8; ++j) {
      float a = __expf(dtv * Ac[j]);
      Q[j] = a * Q[j] + dbx * sB[sb + j];
      P[j] *= a;
    }
    __syncthreads();
  }
  size_t base = (size_t)chunk * NPAIR + d * DS + sb;
#pragma unroll
  for (int j = 0; j < 8; ++j) { Pg[base + j] = P[j]; Qg[base + j] = Q[j]; }
}

// ---------------------------------------------------------------------------
// K6: scan across chunks (one block, 2048 independent pairs, 8 per thread).
// Writes the chunk-entry state h for every chunk.
// ---------------------------------------------------------------------------
__global__ __launch_bounds__(256)
void spa_scan_prefix(const float* __restrict__ Pg, const float* __restrict__ Qg,
                     float* __restrict__ Hst)
{
  const int tid = threadIdx.x;
  float h[8];
#pragma unroll
  for (int j = 0; j < 8; ++j) h[j] = 0.f;
  for (int ch = 0; ch < NCH; ++ch) {
    size_t base = (size_t)ch * NPAIR;
#pragma unroll
    for (int j = 0; j < 8; ++j) {
      size_t q = base + tid + 256 * j;
      Hst[q] = h[j];
      h[j] = Pg[q] * h[j] + Qg[q];
    }
  }
}

// ---------------------------------------------------------------------------
// K7: replay chunks from correct entry state; fuse y = sum_s h*C + D*xc and
// gating with silu(z). Output yg [N, DI].
// ---------------------------------------------------------------------------
__global__ __launch_bounds__(256)
void spa_scan_final(const float* __restrict__ dt, const float* __restrict__ xc,
                    const float* __restrict__ xdb, const float* __restrict__ xz,
                    const float* __restrict__ Alog, const float* __restrict__ Dvec,
                    const float* __restrict__ Hst, float* __restrict__ yg)
{
  __shared__ float sBC[32];
  __shared__ float part[256];
  const int tid = threadIdx.x;
  const int d  = tid & (DI - 1);
  const int sb = (tid >> 7) * 8;
  const int chunk = blockIdx.x;
  float Ac[8], h[8];
  size_t hbase = (size_t)chunk * NPAIR + d * DS + sb;
#pragma unroll
  for (int j = 0; j < 8; ++j) {
    Ac[j] = -__expf(Alog[d * DS + sb + j]);
    h[j] = Hst[hbase + j];
  }
  const float Dd = Dvec[d];
  for (int t = 0; t < CL; ++t) {
    int n = chunk * CL + t;
    if (tid < 32) sBC[tid] = xdb[(size_t)n * 48 + 4 + tid];  // B[0:16], C[0:16]
    __syncthreads();
    float dtv = dt[(size_t)n * DI + d];
    float xcv = xc[(size_t)n * DI + d];
    float dbx = dtv * xcv;
    if (t + 1 < CL) {
      __builtin_prefetch(&dt[(size_t)(n + 1) * DI + d], 0, 1);
      __builtin_prefetch(&xc[(size_t)(n + 1) * DI + d], 0, 1);
    }
    float yp = 0.f;
#pragma unroll
    for (int j = 0; j < 8; ++j) {
      float a = __expf(dtv * Ac[j]);
      h[j] = a * h[j] + dbx * sBC[sb + j];
      yp += h[j] * sBC[16 + sb + j];
    }
    part[tid] = yp;
    __syncthreads();
    if (tid < DI) {   // d == tid here; combine the two s-halves
      float y = part[tid] + part[tid + 128] + Dd * xcv;
      float zv = xz[(size_t)n * 256 + 128 + d];
      yg[(size_t)n * DI + d] = y * silu_f(zv);
    }
    __syncthreads();
  }
}

// ---------------------------------------------------------------------------
// K9: SpeMamba fully fused. 16 threads per sequence (one per d_inner channel),
// 16 sequences per block. State and conv kept in registers; cross-channel
// projections via LDS.
// ---------------------------------------------------------------------------
__global__ __launch_bounds__(256)
void spe_mamba(const float* __restrict__ Xtok,
               const float* __restrict__ inw, const float* __restrict__ cw,
               const float* __restrict__ cb,  const float* __restrict__ xpw,
               const float* __restrict__ dtw, const float* __restrict__ dtb,
               const float* __restrict__ Alog, const float* __restrict__ Dv,
               const float* __restrict__ outw, float* __restrict__ ye)
{
  __shared__ float s_inw[32 * 8];
  __shared__ float s_cw[16 * 4];
  __shared__ float s_cb[16], s_dtw[16], s_dtb[16], s_D[16];
  __shared__ float s_xp[33 * 16];
  __shared__ float s_A[16 * 16];
  __shared__ float s_ow[8 * 16];
  __shared__ float sXt[16 * 64];
  __shared__ float s_xc[16][8][16];
  __shared__ float s_xdb[16][8 * 33];
  __shared__ float s_y[16][8][16];

  const int tid = threadIdx.x;
  s_inw[tid] = inw[tid];                       // 256
  s_A[tid]   = Alog[tid];                      // 256
  if (tid < 64)  s_cw[tid] = cw[tid];
  if (tid < 128) s_ow[tid] = outw[tid];
  if (tid < 16) {
    s_cb[tid] = cb[tid]; s_dtw[tid] = dtw[tid];
    s_dtb[tid] = dtb[tid]; s_D[tid] = Dv[tid];
  }
  for (int i = tid; i < 33 * 16; i += 256) s_xp[i] = xpw[i];
  const int nb = blockIdx.x * 16;
  for (int i = tid; i < 16 * 64; i += 256) sXt[i] = Xtok[(size_t)nb * 64 + i];
  __syncthreads();

  const int sq = tid >> 4;
  const int d  = tid & 15;
  const float* xt = &sXt[sq * 64];

  // in-proj (xi, z) for own channel d
  float xi[8], zz[8];
#pragma unroll
  for (int t = 0; t < 8; ++t) {
    float a = 0.f, b = 0.f;
#pragma unroll
    for (int f = 0; f < 8; ++f) {
      float xv = xt[t * 8 + f];
      a += s_inw[d * 8 + f] * xv;
      b += s_inw[(16 + d) * 8 + f] * xv;
    }
    xi[t] = a; zz[t] = b;
  }
  // causal depthwise conv (k=4) + SiLU, in registers
  float xc[8];
#pragma unroll
  for (int t = 0; t < 8; ++t) {
    float acc = s_cb[d];
#pragma unroll
    for (int j = 0; j < 4; ++j) {
      int tt = t - 3 + j;
      if (tt >= 0) acc += s_cw[d * 4 + j] * xi[tt];
    }
    xc[t] = silu_f(acc);
    s_xc[sq][t][d] = xc[t];
  }
  __syncthreads();
  // x-proj: rows d, d+16, and row 32 handled by d==0
  for (int rr = 0; rr < 3; ++rr) {
    if (rr == 2 && d != 0) break;
    int r = (rr == 0) ? d : (rr == 1 ? d + 16 : 32);
    for (int t = 0; t < 8; ++t) {
      float v = 0.f;
#pragma unroll
      for (int dd = 0; dd < 16; ++dd) v += s_xc[sq][t][dd] * s_xp[r * 16 + dd];
      s_xdb[sq][t * 33 + r] = v;
    }
  }
  __syncthreads();
  // selective scan (L=8) for own channel d
  float Ac[16], h[16];
#pragma unroll
  for (int s = 0; s < 16; ++s) { Ac[s] = -__expf(s_A[d * 16 + s]); h[s] = 0.f; }
#pragma unroll
  for (int t = 0; t < 8; ++t) {
    const float* row = &s_xdb[sq][t * 33];
    float dtv = softplus_f(row[0] * s_dtw[d] + s_dtb[d]);
    float dbx = dtv * xc[t];
    float y = 0.f;
#pragma unroll
    for (int s = 0; s < 16; ++s) {
      float a = __expf(dtv * Ac[s]);
      h[s] = a * h[s] + dbx * row[1 + s];
      y += h[s] * row[17 + s];
    }
    s_y[sq][t][d] = (y + s_D[d] * xc[t]) * silu_f(zz[t]);
  }
  __syncthreads();
  // out-proj: 4 of the 64 (t,f) outputs per thread
#pragma unroll
  for (int k = 0; k < 4; ++k) {
    int idx = d * 4 + k;
    int t = idx >> 3, f = idx & 7;
    float v = 0.f;
#pragma unroll
    for (int dd = 0; dd < 16; ++dd) v += s_y[sq][t][dd] * s_ow[f * 16 + dd];
    ye[(size_t)(nb + sq) * 64 + t * 8 + f] = v;
  }
}

// ---------------------------------------------------------------------------
// K10: deterministic GroupNorm statistics. 32 jobs = {ys,ye} x 4 batches x
// 4 groups; each job tree-reduces 65536 values (no float atomics -> bitwise
// deterministic across graph replays).
// ---------------------------------------------------------------------------
__global__ __launch_bounds__(256)
void gn_stats(const float* __restrict__ ys, const float* __restrict__ ye,
              float* __restrict__ stats)
{
  __shared__ float red[256][2];
  const int job = blockIdx.x;
  const int tensor = job >> 4;
  const int b = (job >> 2) & 3;
  const int g = job & 3;
  const float* T = tensor ? ye : ys;
  const int tid = threadIdx.x;
  float s = 0.f, ss = 0.f;
  for (int i = tid; i < 65536; i += 256) {
    int n = b * 4096 + (i >> 4);
    int c = g * 16 + (i & 15);
    float v = T[(size_t)n * 64 + c];
    s += v; ss += v * v;
  }
  red[tid][0] = s; red[tid][1] = ss;
  __syncthreads();
  for (int off = 128; off > 0; off >>= 1) {
    if (tid < off) {
      red[tid][0] += red[tid + off][0];
      red[tid][1] += red[tid + off][1];
    }
    __syncthreads();
  }
  if (tid == 0) { stats[job * 2] = red[0][0]; stats[job * 2 + 1] = red[0][1]; }
}

// ---------------------------------------------------------------------------
// K11: fused GN-normalize + SiLU + residuals + softmax(att) combine.
// ---------------------------------------------------------------------------
__global__ void final_combine(const float* __restrict__ x,
                              const float* __restrict__ ys,
                              const float* __restrict__ ye,
                              const float* __restrict__ stats,
                              const float* __restrict__ gwa, const float* __restrict__ gba,
                              const float* __restrict__ gwe, const float* __restrict__ gbe,
                              const float* __restrict__ att,
                              float* __restrict__ out)
{
  int i = blockIdx.x * blockDim.x + threadIdx.x;   // NTOK*64
  if (i >= NTOK * 64) return;
  int c = i & 63, n = i >> 6;
  int b = n >> 12, g = c >> 4;
  const float cnt = 65536.f, eps = 1e-5f;
  int ja = (0 * 4 + b) * 4 + g;
  int je = (1 * 4 + b) * 4 + g;
  float mua = stats[ja * 2] / cnt;
  float va  = stats[ja * 2 + 1] / cnt - mua * mua;
  float ra  = rsqrtf(va + eps);
  float mue = stats[je * 2] / cnt;
  float ve  = stats[je * 2 + 1] / cnt - mue * mue;
  float re  = rsqrtf(ve + eps);
  float sa = silu_f((ys[i] - mua) * ra * gwa[c] + gba[c]);
  float se = silu_f((ye[i] - mue) * re * gwe[c] + gbe[c]);
  size_t xidx = (size_t)b * 262144 + (size_t)c * 4096 + (size_t)(n & 4095);
  float xv = x[xidx];
  float a0 = att[0], a1 = att[1];
  float m  = fmaxf(a0, a1);
  float e0 = __expf(a0 - m), e1 = __expf(a1 - m);
  float w0 = e0 / (e0 + e1), w1 = e1 / (e0 + e1);
  out[xidx] = w0 * (sa + xv) + w1 * (xv + se) + xv;
}

// ---------------------------------------------------------------------------
extern "C" void kernel_launch(void* const* d_in, const int* in_sizes, int n_in,
                              void* d_out, int out_size, void* d_ws, size_t ws_size,
                              hipStream_t stream)
{
  (void)in_sizes; (void)n_in; (void)out_size; (void)ws_size;
  const float* x        = (const float*)d_in[0];
  const float* spa_inw  = (const float*)d_in[1];
  const float* spa_cw   = (const float*)d_in[2];
  const float* spa_cb   = (const float*)d_in[3];
  const float* spa_xpw  = (const float*)d_in[4];
  const float* spa_dtw  = (const float*)d_in[5];
  const float* spa_dtb  = (const float*)d_in[6];
  const float* spa_Al   = (const float*)d_in[7];
  const float* spa_Dv   = (const float*)d_in[8];
  const float* spa_ow   = (const float*)d_in[9];
  const float* spe_inw  = (const float*)d_in[10];
  const float* spe_cw   = (const float*)d_in[11];
  const float* spe_cb   = (const float*)d_in[12];
  const float* spe_xpw  = (const float*)d_in[13];
  const float* spe_dtw  = (const float*)d_in[14];
  const float* spe_dtb  = (const float*)d_in[15];
  const float* spe_Al   = (const float*)d_in[16];
  const float* spe_Dv   = (const float*)d_in[17];
  const float* spe_ow   = (const float*)d_in[18];
  const float* gwa      = (const float*)d_in[19];
  const float* gba      = (const float*)d_in[20];
  const float* gwe      = (const float*)d_in[21];
  const float* gbe      = (const float*)d_in[22];
  const float* att      = (const float*)d_in[23];
  float* out = (float*)d_out;

  // Workspace carve-up (floats).
  float* w    = (float*)d_ws;
  float* Xtok = w;  w += (size_t)NTOK * CCH;     // 1.0M
  float* xz   = w;  w += (size_t)NTOK * 256;     // 4.2M
  float* xc   = w;  w += (size_t)NTOK * DI;      // 2.1M
  float* xdb  = w;  w += (size_t)NTOK * 48;      // 0.8M
  float* dt   = w;  w += (size_t)NTOK * DI;      // 2.1M
  float* Pg   = w;  w += (size_t)NCH * NPAIR;    // 0.5M
  float* Qg   = w;  w += (size_t)NCH * NPAIR;    // 0.5M
  float* Hst  = w;  w += (size_t)NCH * NPAIR;    // 0.5M
  float* yg   = w;  w += (size_t)NTOK * DI;      // 2.1M
  float* ys   = w;  w += (size_t)NTOK * CCH;     // 1.0M
  float* ye   = w;  w += (size_t)NTOK * CCH;     // 1.0M
  float* stats = w; w += 64;                     // ~64 MB total

  // --- SpaMamba ---
  repack_x_tdm<<<NTOK / 64, 256, 0, stream>>>(x, Xtok);
  gemm_xw_t<64, 256, 256><<<NTOK / 128, 256, 0, stream>>>(Xtok, spa_inw, xz);
  spa_conv_silu<<<(NTOK * DI + 255) / 256, 256, 0, stream>>>(xz, spa_cw, spa_cb, xc);
  gemm_xw_t<128, 48, 36><<<NTOK / 128, 256, 0, stream>>>(xc, spa_xpw, xdb);
  spa_dt<<<(NTOK * DI + 255) / 256, 256, 0, stream>>>(xdb, spa_dtw, spa_dtb, dt);
  spa_scan_chunk<<<NCH, 256, 0, stream>>>(dt, xc, xdb, spa_Al, Pg, Qg);
  spa_scan_prefix<<<1, 256, 0, stream>>>(Pg, Qg, Hst);
  spa_scan_final<<<NCH, 256, 0, stream>>>(dt, xc, xdb, xz, spa_Al, spa_Dv, Hst, yg);
  gemm_xw_t<128, 64, 64><<<NTOK / 128, 256, 0, stream>>>(yg, spa_ow, ys);

  // --- SpeMamba (fully fused) ---
  spe_mamba<<<NTOK / 16, 256, 0, stream>>>(Xtok, spe_inw, spe_cw, spe_cb, spe_xpw,
                                           spe_dtw, spe_dtb, spe_Al, spe_Dv,
                                           spe_ow, ye);

  // --- GroupNorms + combine ---
  gn_stats<<<32, 256, 0, stream>>>(ys, ye, stats);
  final_combine<<<(NTOK * 64 + 255) / 256, 256, 0, stream>>>(
      x, ys, ye, stats, gwa, gba, gwe, gbe, att, out);
}
